// MultiHeadAttention_7052336300101
// MI455X (gfx1250) — matrix operand order repaired and checked by reading, not compiled, not run on hardware
//
#include <hip/hip_runtime.h>

// Problem constants (match reference)
#define B_  4
#define S_  2048
#define E_  1024
#define H_  16
#define D_  64
#define BS_ (B_ * S_)   // 8192

typedef __attribute__((ext_vector_type(16))) _Float16 v16h;
typedef __attribute__((ext_vector_type(8)))  float    v8f;
typedef int v4i_ __attribute__((vector_size(4 * sizeof(int))));

union FragA { v16h v; uint4 q[2]; };

__device__ __forceinline__ v8f wmma_f16(const FragA& a, const FragA& b, v8f c) {
  // D = A(16x32 f16) * B(32x16 f16) + C(16x16 f32)
  return __builtin_amdgcn_wmma_f32_16x16x32_f16(false, a.v, false, b.v,
                                                (short)0, c, false, false);
}

// Async global->LDS 16B copy (GLOBAL_LOAD_ASYNC_TO_LDS_B128, ASYNCcnt).
__device__ __forceinline__ void async_b128(void* lds_dst, const void* gsrc) {
  __builtin_amdgcn_global_load_async_to_lds_b128(
      (__attribute__((address_space(1))) v4i_*)(uintptr_t)gsrc,
      (__attribute__((address_space(3))) v4i_*)(uintptr_t)lds_dst,
      0, 0);
}
__device__ __forceinline__ void wait_async0() {
  asm volatile("s_wait_asynccnt 0x0" ::: "memory");
}

//======================================================================
// Pre-pass A: convert activations f32 -> f16 (contiguous).
//======================================================================
__global__ void __launch_bounds__(256)
cvt_x_kernel(const float* __restrict__ xq, const float* __restrict__ xk,
             const float* __restrict__ xv,
             _Float16* __restrict__ oq, _Float16* __restrict__ ok,
             _Float16* __restrict__ ov)
{
  const float* src; _Float16* dst;
  if (blockIdx.z == 0)      { src = xq; dst = oq; }
  else if (blockIdx.z == 1) { src = xk; dst = ok; }
  else                      { src = xv; dst = ov; }
  const size_t i = ((size_t)blockIdx.x * 256 + threadIdx.x) * 8;
  float4 f0 = *(const float4*)(src + i);
  float4 f1 = *(const float4*)(src + i + 4);
  union { _Float16 h[8]; uint4 u; } p;
  p.h[0] = (_Float16)f0.x; p.h[1] = (_Float16)f0.y;
  p.h[2] = (_Float16)f0.z; p.h[3] = (_Float16)f0.w;
  p.h[4] = (_Float16)f1.x; p.h[5] = (_Float16)f1.y;
  p.h[6] = (_Float16)f1.z; p.h[7] = (_Float16)f1.w;
  *(uint4*)(dst + i) = p.u;
}

//======================================================================
// Pre-pass B: weights f32 [k][n] -> f16 transposed [n][k] (LDS tile).
//======================================================================
__global__ void __launch_bounds__(256)
cvt_w_kernel(const float* __restrict__ wq, const float* __restrict__ wk,
             const float* __restrict__ wv, const float* __restrict__ wo,
             _Float16* __restrict__ tq, _Float16* __restrict__ tk,
             _Float16* __restrict__ tv, _Float16* __restrict__ tw)
{
  __shared__ _Float16 T[64 * 64];
  const float* src; _Float16* dst;
  if (blockIdx.z == 0)      { src = wq; dst = tq; }
  else if (blockIdx.z == 1) { src = wk; dst = tk; }
  else if (blockIdx.z == 2) { src = wv; dst = tv; }
  else                      { src = wo; dst = tw; }
  const int k0 = blockIdx.x * 64, n0 = blockIdx.y * 64;
  const int r = threadIdx.x >> 2, c = (threadIdx.x & 3) * 16;
  const float* s = src + (size_t)(k0 + r) * E_ + n0 + c;
  #pragma unroll
  for (int i = 0; i < 16; i += 4) {
    float4 f = *(const float4*)(s + i);
    T[(c + i + 0) * 64 + r] = (_Float16)f.x;
    T[(c + i + 1) * 64 + r] = (_Float16)f.y;
    T[(c + i + 2) * 64 + r] = (_Float16)f.z;
    T[(c + i + 3) * 64 + r] = (_Float16)f.w;
  }
  __syncthreads();
  _Float16* o = dst + (size_t)(n0 + r) * E_ + k0 + c;
  *(uint4*)(o)     = *(const uint4*)&T[r * 64 + c];
  *(uint4*)(o + 8) = *(const uint4*)&T[r * 64 + c + 8];
}

//======================================================================
// Async tile staging for the projection GEMMs.
// A: 256x32 f16 (activations, row-major), B: 64x32 f16 (wT rows).
//======================================================================
__device__ __forceinline__ void stage_gemm_async(
    _Float16* AsBuf, _Float16* BsBuf,
    const _Float16* __restrict__ x, const _Float16* __restrict__ wT,
    int m0, int n0, int k0, int tid)
{
  #pragma unroll
  for (int i = 0; i < 4; ++i) {          // A: 1024 x 16B chunks
    const int c = i * 256 + tid;
    const int r = c >> 2, co = (c & 3) * 8;
    async_b128(AsBuf + r * 32 + co, x + (size_t)(m0 + r) * E_ + k0 + co);
  }
  {                                      // B: 256 x 16B chunks
    const int r = tid >> 2, co = (tid & 3) * 8;
    async_b128(BsBuf + r * 32 + co, wT + (size_t)(n0 + r) * E_ + k0 + co);
  }
}

//======================================================================
// Shared GEMM inner step: 2 A-frags x 4 B-frags -> 8 WMMAs, loads batched
// ahead of the WMMA chain for a single dscnt wait + co-executing WMMAs.
//======================================================================
__device__ __forceinline__ void gemm_step(
    const _Float16* Asb, const _Float16* Bsb,
    int wid, int hl, int l15, v8f acc[2][4])
{
  FragA a[2];
  const int kA = hl * 8;
  #pragma unroll
  for (int s = 0; s < 2; ++s) {
    const int arow = wid * 32 + s * 16 + l15;
    a[s].q[0] = *(const uint4*)&Asb[arow * 32 + kA];
    a[s].q[1] = *(const uint4*)&Asb[arow * 32 + kA + 16];
  }
  FragA bf[4];
  const int kB = hl * 8;
  #pragma unroll
  for (int c = 0; c < 4; ++c) {
    const int bn = c * 16 + l15;
    bf[c].q[0] = *(const uint4*)&Bsb[bn * 32 + kB];
    bf[c].q[1] = *(const uint4*)&Bsb[bn * 32 + kB + 16];
  }
  #pragma unroll
  for (int c = 0; c < 4; ++c)
    #pragma unroll
    for (int s = 0; s < 2; ++s)
      acc[s][c] = wmma_f16(a[s], bf[c], acc[s][c]);
}

//======================================================================
// Kernel 1: fused QKV projection (all-f16 tiles, async double-buffered).
// Block tile 256x64, 8 waves x (32x64). Q gets 0.125 folded in; V is
// stored transposed [B,H,D,S].
//======================================================================
__global__ void __launch_bounds__(256)
qkv_kernel(const _Float16* __restrict__ xq, const _Float16* __restrict__ xk,
           const _Float16* __restrict__ xv,
           const _Float16* __restrict__ wqT, const _Float16* __restrict__ wkT,
           const _Float16* __restrict__ wvT,
           const float* __restrict__ bq, const float* __restrict__ bk,
           const float* __restrict__ bv,
           _Float16* __restrict__ qh, _Float16* __restrict__ kh,
           _Float16* __restrict__ vhT)
{
  __shared__ _Float16 As[2][256 * 32];
  __shared__ _Float16 Bs[2][64 * 32];

  const _Float16* x; const _Float16* wT; const float* bias; _Float16* dst;
  float scale = 1.0f;
  const int which = blockIdx.z;
  if (which == 0)      { x = xq; wT = wqT; bias = bq; dst = qh;  scale = 0.125f; }
  else if (which == 1) { x = xk; wT = wkT; bias = bk; dst = kh; }
  else                 { x = xv; wT = wvT; bias = bv; dst = vhT; }

  const int m0 = blockIdx.x * 256;
  const int n0 = blockIdx.y * 64;
  const int tid = threadIdx.x;
  const int wid = tid >> 5, lane = tid & 31;
  const int hl = lane >> 4, l15 = lane & 15;

  v8f acc[2][4] = {};
  stage_gemm_async(As[0], Bs[0], x, wT, m0, n0, 0, tid);

  int buf = 0;
  for (int k0 = 0; k0 < E_; k0 += 32) {
    wait_async0();
    __syncthreads();
    if (k0 + 32 < E_)
      stage_gemm_async(As[buf ^ 1], Bs[buf ^ 1], x, wT, m0, n0, k0 + 32, tid);
    gemm_step(As[buf], Bs[buf], wid, hl, l15, acc);
    buf ^= 1;
  }

  // Epilogue: + bias, * scale; Q/K -> [B,H,S,D], V -> [B,H,D,S]
  #pragma unroll
  for (int c = 0; c < 4; ++c) {
    const int n = n0 + c * 16 + l15;
    const int hh = n >> 6, dd = n & 63;
    const float bvl = bias[n];
    #pragma unroll
    for (int s = 0; s < 2; ++s) {
      #pragma unroll
      for (int j = 0; j < 8; ++j) {
        const int m = m0 + wid * 32 + s * 16 + j + hl * 8;
        const int bb = m >> 11, ss = m & (S_ - 1);
        const _Float16 val = (_Float16)((acc[s][c][j] + bvl) * scale);
        if (which == 2)
          dst[(((size_t)bb * H_ + hh) * D_ + dd) * S_ + ss] = val;
        else
          dst[(((size_t)bb * H_ + hh) * S_ + ss) * D_ + dd] = val;
      }
    }
  }
}

//======================================================================
// Kernel 2: causal flash attention, async double-buffered K/V tiles.
//======================================================================
__global__ void __launch_bounds__(256)
attn_kernel(const _Float16* __restrict__ qg, const _Float16* __restrict__ kg,
            const _Float16* __restrict__ vgT, _Float16* __restrict__ aout)
{
  const int bh = blockIdx.y;              // 0..63
  const int bb = bh >> 4, hh = bh & 15;
  const int q0 = blockIdx.x * 128;

  const _Float16* qbase = qg + ((size_t)bh * S_ + q0) * D_;
  const _Float16* kbase = kg + (size_t)bh * S_ * D_;
  const _Float16* vtbase = vgT + (size_t)bh * D_ * S_;   // [dim][key]

  __shared__ _Float16 Qs[128 * 64];       // [qrow][dim]
  __shared__ _Float16 Ks[2][64 * 64];     // [key][dim]
  __shared__ _Float16 Vt[2][64 * 64];     // [dim][key]
  __shared__ _Float16 Ps[8][16 * 64];     // per-wave P scratch

  const int tid = threadIdx.x;
  const int wid = tid >> 5, lane = tid & 31;
  const int hl = lane >> 4, l15 = lane & 15;

  // Q tile (16KB) + first K/V tiles: all async
  #pragma unroll
  for (int i = 0; i < 4; ++i) {
    const int c = i * 256 + tid;
    const int r = c >> 3, co = (c & 7) * 8;
    async_b128(Qs + r * 64 + co, qbase + (size_t)r * 64 + co);
  }
  const int kr = tid >> 3, kco = (tid & 7) * 8;  // covers rows 0..31
  {
    async_b128(Ks[0] + kr * 64 + kco,        kbase + (size_t)kr * 64 + kco);
    async_b128(Ks[0] + (kr + 32) * 64 + kco, kbase + (size_t)(kr + 32) * 64 + kco);
    async_b128(Vt[0] + kr * 64 + kco,        vtbase + (size_t)kr * S_ + kco);
    async_b128(Vt[0] + (kr + 32) * 64 + kco, vtbase + (size_t)(kr + 32) * S_ + kco);
  }
  wait_async0();
  __syncthreads();

  // Preload this wave's Q A-fragments (rows wid*16..+15, dims 0..63)
  FragA qa[2];
  const int qrowL = wid * 16 + l15;
  #pragma unroll
  for (int f = 0; f < 2; ++f) {
    const int kA = f * 32 + hl * 8;
    qa[f].q[0] = *(const uint4*)&Qs[qrowL * 64 + kA];
    qa[f].q[1] = *(const uint4*)&Qs[qrowL * 64 + kA + 16];
  }

  v8f acc[4] = {};
  float mrow[8], lrow[8];
  #pragma unroll
  for (int j = 0; j < 8; ++j) { mrow[j] = -1e30f; lrow[j] = 0.0f; }

  const int ktiles = (q0 + 128) >> 6;   // causal: keys < q0+128
  int buf = 0;
  for (int kt = 0; kt < ktiles; ++kt) {
    const int kk0 = kt * 64;
    if (kt) { wait_async0(); __syncthreads(); }
    if (kt + 1 < ktiles) {
      const int nk0 = kk0 + 64;
      _Float16* Kn = Ks[buf ^ 1]; _Float16* Vn = Vt[buf ^ 1];
      async_b128(Kn + kr * 64 + kco,        kbase + (size_t)(nk0 + kr) * 64 + kco);
      async_b128(Kn + (kr + 32) * 64 + kco, kbase + (size_t)(nk0 + kr + 32) * 64 + kco);
      async_b128(Vn + kr * 64 + kco,        vtbase + (size_t)kr * S_ + nk0 + kco);
      async_b128(Vn + (kr + 32) * 64 + kco, vtbase + (size_t)(kr + 32) * S_ + nk0 + kco);
    }
    const _Float16* Kb = Ks[buf];
    const _Float16* Vb = Vt[buf];

    // ---- scores = Q * K^T (batched B-frag loads, back-to-back WMMAs) ----
    v8f sc4[4] = {};
    #pragma unroll
    for (int f = 0; f < 2; ++f) {
      FragA bf[4];
      const int kB = f * 32 + hl * 8;
      #pragma unroll
      for (int c = 0; c < 4; ++c) {
        const int bn = c * 16 + l15;
        bf[c].q[0] = *(const uint4*)&Kb[bn * 64 + kB];
        bf[c].q[1] = *(const uint4*)&Kb[bn * 64 + kB + 16];
      }
      #pragma unroll
      for (int c = 0; c < 4; ++c) sc4[c] = wmma_f16(qa[f], bf[c], sc4[c]);
    }

    // ---- causal mask + row max ----
    float rmax[8];
    #pragma unroll
    for (int j = 0; j < 8; ++j) rmax[j] = -1e30f;
    #pragma unroll
    for (int c = 0; c < 4; ++c) {
      const int kabs = kk0 + c * 16 + l15;
      #pragma unroll
      for (int j = 0; j < 8; ++j) {
        const int qabs = q0 + wid * 16 + j + hl * 8;
        float s = sc4[c][j];
        s = (kabs <= qabs) ? s : -1e30f;
        sc4[c][j] = s;
        rmax[j] = fmaxf(rmax[j], s);
      }
    }
    #pragma unroll
    for (int j = 0; j < 8; ++j) {   // reduce over the 16 lanes of each half
      float v = rmax[j];
      v = fmaxf(v, __shfl_xor(v, 1, 32));
      v = fmaxf(v, __shfl_xor(v, 2, 32));
      v = fmaxf(v, __shfl_xor(v, 4, 32));
      v = fmaxf(v, __shfl_xor(v, 8, 32));
      rmax[j] = v;
    }

    // ---- online softmax update ----
    #pragma unroll
    for (int j = 0; j < 8; ++j) {
      const float newm = fmaxf(mrow[j], rmax[j]);
      const float alpha = __expf(mrow[j] - newm);
      float rs = 0.0f;
      #pragma unroll
      for (int c = 0; c < 4; ++c) {
        const float p = __expf(sc4[c][j] - newm);
        sc4[c][j] = p;
        rs += p;
      }
      rs += __shfl_xor(rs, 1, 32);
      rs += __shfl_xor(rs, 2, 32);
      rs += __shfl_xor(rs, 4, 32);
      rs += __shfl_xor(rs, 8, 32);
      lrow[j] = lrow[j] * alpha + rs;
      mrow[j] = newm;
      #pragma unroll
      for (int c = 0; c < 4; ++c) acc[c][j] *= alpha;
    }

    // ---- P: C-layout -> A-layout via per-wave LDS ----
    _Float16* P = Ps[wid];
    #pragma unroll
    for (int c = 0; c < 4; ++c)
      #pragma unroll
      for (int j = 0; j < 8; ++j)
        P[(j + hl * 8) * 64 + c * 16 + l15] = (_Float16)sc4[c][j];
    asm volatile("s_wait_dscnt 0x0" ::: "memory");  // same-wave LDS RAW

    // ---- acc += P * V (batched loads) ----
    #pragma unroll
    for (int f = 0; f < 2; ++f) {
      FragA pa;
      const int kA = f * 32 + hl * 8;
      pa.q[0] = *(const uint4*)&P[l15 * 64 + kA];
      pa.q[1] = *(const uint4*)&P[l15 * 64 + kA + 16];
      FragA vb[4];
      const int kB = f * 32 + hl * 8;
      #pragma unroll
      for (int c = 0; c < 4; ++c) {
        const int bn = c * 16 + l15;
        vb[c].q[0] = *(const uint4*)&Vb[bn * 64 + kB];
        vb[c].q[1] = *(const uint4*)&Vb[bn * 64 + kB + 16];
      }
      #pragma unroll
      for (int c = 0; c < 4; ++c) acc[c] = wmma_f16(pa, vb[c], acc[c]);
    }
    buf ^= 1;
  }

  // ---- normalize and store f16 into [B,S,H*D] ----
  #pragma unroll
  for (int c = 0; c < 4; ++c) {
    #pragma unroll
    for (int j = 0; j < 8; ++j) {
      const int qabs = q0 + wid * 16 + j + hl * 8;
      const size_t row = (size_t)bb * S_ + qabs;
      const int col = hh * 64 + c * 16 + l15;
      aout[row * E_ + col] = (_Float16)(acc[c][j] / lrow[j]);
    }
  }
}

//======================================================================
// Kernel 3: output projection (f16 A from ws, f16 woT, f32 out + bias).
// Block tile 256x64 like kernel 1.
//======================================================================
__global__ void __launch_bounds__(256)
out_proj_kernel(const _Float16* __restrict__ a, const _Float16* __restrict__ wT,
                const float* __restrict__ bias, float* __restrict__ out)
{
  __shared__ _Float16 As[2][256 * 32];
  __shared__ _Float16 Bs[2][64 * 32];

  const int m0 = blockIdx.x * 256;
  const int n0 = blockIdx.y * 64;
  const int tid = threadIdx.x;
  const int wid = tid >> 5, lane = tid & 31;
  const int hl = lane >> 4, l15 = lane & 15;

  v8f acc[2][4] = {};
  stage_gemm_async(As[0], Bs[0], a, wT, m0, n0, 0, tid);

  int buf = 0;
  for (int k0 = 0; k0 < E_; k0 += 32) {
    wait_async0();
    __syncthreads();
    if (k0 + 32 < E_)
      stage_gemm_async(As[buf ^ 1], Bs[buf ^ 1], a, wT, m0, n0, k0 + 32, tid);
    gemm_step(As[buf], Bs[buf], wid, hl, l15, acc);
    buf ^= 1;
  }

  #pragma unroll
  for (int c = 0; c < 4; ++c) {
    const int n = n0 + c * 16 + l15;
    const float bvl = bias[n];
    #pragma unroll
    for (int s = 0; s < 2; ++s) {
      #pragma unroll
      for (int j = 0; j < 8; ++j) {
        const int m = m0 + wid * 32 + s * 16 + j + hl * 8;
        out[(size_t)m * E_ + n] = acc[s][c][j] + bvl;
      }
    }
  }
}

//======================================================================
extern "C" void kernel_launch(void* const* d_in, const int* in_sizes, int n_in,
                              void* d_out, int out_size, void* d_ws, size_t ws_size,
                              hipStream_t stream)
{
  (void)in_sizes; (void)n_in; (void)out_size; (void)ws_size;

  const float* q  = (const float*)d_in[0];
  const float* k  = (const float*)d_in[1];
  const float* v  = (const float*)d_in[2];
  // d_in[3] = mask (bool) — causal mask computed analytically, unused
  const float* wq = (const float*)d_in[4];
  const float* bq = (const float*)d_in[5];
  const float* wk = (const float*)d_in[6];
  const float* bk = (const float*)d_in[7];
  const float* wv = (const float*)d_in[8];
  const float* bv = (const float*)d_in[9];
  const float* wo = (const float*)d_in[10];
  const float* bo = (const float*)d_in[11];
  float* out = (float*)d_out;

  const size_t XE = (size_t)BS_ * E_;   // 8M elems
  const size_t WE = (size_t)E_ * E_;    // 1M elems

  _Float16* p = (_Float16*)d_ws;
  _Float16* xq16 = p; p += XE;
  _Float16* xk16 = p; p += XE;
  _Float16* xv16 = p; p += XE;
  _Float16* wqT  = p; p += WE;
  _Float16* wkT  = p; p += WE;
  _Float16* wvT  = p; p += WE;
  _Float16* woT  = p; p += WE;
  _Float16* qh   = p; p += XE;
  _Float16* kh   = p; p += XE;
  _Float16* vhT  = p; p += XE;
  _Float16* at   = p; p += XE;

  cvt_x_kernel<<<dim3((unsigned)(XE / (256 * 8)), 1, 3), 256, 0, stream>>>(
      q, k, v, xq16, xk16, xv16);
  cvt_w_kernel<<<dim3(16, 16, 4), 256, 0, stream>>>(
      wq, wk, wv, wo, wqT, wkT, wvT, woT);
  qkv_kernel<<<dim3(BS_ / 256, E_ / 64, 3), 256, 0, stream>>>(
      xq16, xk16, xv16, wqT, wkT, wvT, bq, bk, bv, qh, kh, vhT);
  attn_kernel<<<dim3(S_ / 128, B_ * H_), 256, 0, stream>>>(qh, kh, vhT, at);
  out_proj_kernel<<<dim3(BS_ / 256, E_ / 64), 256, 0, stream>>>(at, woT, bo, out);
}
